// TwoDimensionalGRU_6305011990599
// MI455X (gfx1250) — matrix operand order repaired and checked by reading, not compiled, not run on hardware
//
#include <hip/hip_runtime.h>

// ---------------------------------------------------------------------------
// 2-D GRU for MI455X (gfx1250): bf16 WMMA (16x16x32, f32 accum) everywhere.
// Phase 1: gx = (x@We^T+be)@W_ih^T + b_ih   (parallel, 1024 blocks)
// Phase 2: persistent column-wavefront kernel, 32 blocks, spin-sync on
//          per-column progress counters (agent-scope acquire/release).
// ---------------------------------------------------------------------------

typedef __attribute__((ext_vector_type(16))) __bf16 v16bf;
typedef __attribute__((ext_vector_type(8)))  __bf16 v8bf;
typedef __attribute__((ext_vector_type(8)))  float  v8f;

__device__ __forceinline__ v8f wmma_bf16(v16bf a, v16bf b, v8f c) {
  // emits v_wmma_f32_16x16x32_bf16
  return __builtin_amdgcn_wmma_f32_16x16x32_bf16(false, a, false, b, (short)0, c,
                                                 false, false);
}

// A operand: 16x32 bf16 tile, row-major, leading dim ldk.
// ISA layout: lane l -> row l&15; elems 0..7 = K g*8..g*8+7, elems 8..15 =
// K 16+g*8..+7 (g = lane>>4). Two aligned 16-byte loads per lane.
__device__ __forceinline__ v16bf load_a16x32(const __bf16* base, int ldk, int lane) {
  const int m = lane & 15;
  const int g = lane >> 4;
  const __bf16* p = base + m * ldk + g * 8;
  v8bf lo = *(const v8bf*)(p);          // K = g*8 .. +7
  v8bf hi = *(const v8bf*)(p + 16);     // K = 16+g*8 .. +7
  v16bf r;
#pragma unroll
  for (int e = 0; e < 8; ++e) { r[e] = lo[e]; r[e + 8] = hi[e]; }
  return r;
}

// B operand: rows K=k0..k0+31, cols n0..n0+15 of B = W^T, where W is the
// original row-major [N x K] weight matrix (so no transpose is ever needed).
// ISA layout: lane l -> col l&15; elems 0..7 = K g*8..g*8+7, elems 8..15 =
// K 16+g*8..+7 (g = lane>>4).
__device__ __forceinline__ v16bf load_b32x16(const __bf16* W, int ldk, int n0,
                                             int k0, int lane) {
  const int n = lane & 15;
  const int g = lane >> 4;
  const __bf16* p = W + (size_t)(n0 + n) * ldk + k0 + g * 8;
  v8bf lo = *(const v8bf*)(p);          // K = g*8 .. +7
  v8bf hi = *(const v8bf*)(p + 16);     // K = 16+g*8 .. +7
  v16bf r;
#pragma unroll
  for (int e = 0; e < 8; ++e) { r[e] = lo[e]; r[e + 8] = hi[e]; }
  return r;
}

__device__ __forceinline__ float sigmoidf_(float v) {
  return 1.0f / (1.0f + __expf(-v));
}

// --------------------------- weight conversion -----------------------------
__global__ void convert_bf16_kernel(const float* __restrict__ src,
                                    __bf16* __restrict__ dst, int n) {
  int i = blockIdx.x * blockDim.x + threadIdx.x;
  if (i < n) dst[i] = (__bf16)src[i];
}

// ------------------------- phase 1: embed + gx -----------------------------
// one block per cell; 256 threads = 8 waves. Output gx[cell][b][o] bf16.
__global__ __launch_bounds__(256)
void gx_kernel(const float* __restrict__ x, const float* __restrict__ be,
               const float* __restrict__ b_ih, const __bf16* __restrict__ We_bf,
               const __bf16* __restrict__ Wih_bf, __bf16* __restrict__ gx_all) {
  __shared__ __align__(16) __bf16 ldsX[64 * 64];    //  8 KB: x tile bf16
  __shared__ __align__(16) __bf16 ldsE[64 * 128];   // 16 KB: emb bf16
  const int cell = blockIdx.x;                       // i*32 + j
  const float* xc = x + (size_t)cell * 64;           // x[b][i][j][:] stride 65536

  // stage + convert x tile (64 batches x 64 feats)
  for (int t = threadIdx.x; t < 64 * 16; t += 256) {
    int b = t >> 4, d = (t & 15) * 4;
    float4 v = *(const float4*)(xc + (size_t)b * (32 * 32 * 64) + d);
    __bf16* q = &ldsX[b * 64 + d];
    q[0] = (__bf16)v.x; q[1] = (__bf16)v.y; q[2] = (__bf16)v.z; q[3] = (__bf16)v.w;
  }
  __syncthreads();

  const int wave = threadIdx.x >> 5, lane = threadIdx.x & 31;

  // emb = x @ We^T + be : M=64 N=128 K=64 -> 32 tiles, 4 per wave
#pragma unroll
  for (int q = 0; q < 4; ++q) {
    int t = wave * 4 + q;
    int mt = t & 3, nt = t >> 2;
    v8f acc = {};
#pragma unroll
    for (int kk = 0; kk < 64; kk += 32) {
      v16bf a = load_a16x32(ldsX + mt * 16 * 64 + kk, 64, lane);
      v16bf b = load_b32x16(We_bf, 64, nt * 16, kk, lane);
      acc = wmma_bf16(a, b, acc);
    }
    int row0 = mt * 16 + (lane >> 4) * 8;
    int col  = nt * 16 + (lane & 15);
    float bias = be[col];
#pragma unroll
    for (int v = 0; v < 8; ++v)
      ldsE[(row0 + v) * 128 + col] = (__bf16)(acc[v] + bias);
  }
  __syncthreads();

  // gx = emb @ W_ih^T + b_ih : M=64 N=768 K=128 -> 192 tiles, 24 per wave
  __bf16* gxc = gx_all + (size_t)cell * 64 * 768;
  for (int q = 0; q < 24; ++q) {
    int t = wave * 24 + q;
    int mt = t & 3, nt = t >> 2;
    v8f acc = {};
#pragma unroll
    for (int kk = 0; kk < 128; kk += 32) {
      v16bf a = load_a16x32(ldsE + mt * 16 * 128 + kk, 128, lane);
      v16bf b = load_b32x16(Wih_bf, 128, nt * 16, kk, lane);
      acc = wmma_bf16(a, b, acc);
    }
    int row0 = mt * 16 + (lane >> 4) * 8;
    int col  = nt * 16 + (lane & 15);
    float bias = b_ih[col];
#pragma unroll
    for (int v = 0; v < 8; ++v)
      gxc[(size_t)(row0 + v) * 768 + col] = (__bf16)(acc[v] + bias);
  }
}

// ---------------------- phase 2: persistent wavefront ----------------------
// 32 blocks, block j owns grid column j and walks rows i = 0..31,
// spin-waiting until column j-1 has finished row i (progress[j-1] > i).
__global__ __launch_bounds__(256)
void wavefront_kernel(const __bf16* __restrict__ gx_all,
                      const float* __restrict__ b_hh,
                      const float* __restrict__ bc,
                      const __bf16* __restrict__ Whh_bf,
                      const __bf16* __restrict__ Wc_bf,
                      __bf16* __restrict__ hbuf,
                      int* __restrict__ progress,
                      float* __restrict__ out) {
  __shared__ __align__(16) __bf16 ldsH [64 * 256];  // 32 KB  h  = [row | col]
  __shared__ __align__(16) __bf16 ldsHN[64 * 256];  // 32 KB  h_new
  const int j = blockIdx.x;
  const int wave = threadIdx.x >> 5, lane = threadIdx.x & 31;

  for (int i = 0; i < 32; ++i) {
    if (j > 0) {                                     // wait for cell (i, j-1)
      if (threadIdx.x == 0) {
        while (__hip_atomic_load(progress + (j - 1), __ATOMIC_ACQUIRE,
                                 __HIP_MEMORY_SCOPE_AGENT) <= i) {
          __builtin_amdgcn_s_sleep(1);
        }
      }
      __syncthreads();
      __builtin_amdgcn_fence(__ATOMIC_ACQUIRE, "agent");  // invalidate stale L0
    }
    const int cell = i * 32 + j;
    const __bf16* hr = (i > 0) ? hbuf + (size_t)(cell - 32) * 64 * 128 : nullptr;
    const __bf16* hc = (j > 0) ? hbuf + (size_t)(cell - 1)  * 64 * 128 : nullptr;

    // stage h = concat(last_row, last_col) : 64 x 256 bf16 (16-byte copies)
    const uint4 z4 = make_uint4(0u, 0u, 0u, 0u);
    for (int t = threadIdx.x; t < 64 * 16; t += 256) {
      int b = t >> 4, c = (t & 15) * 8;
      *(uint4*)&ldsH[b * 256 + c]       = hr ? *(const uint4*)&hr[b * 128 + c] : z4;
      *(uint4*)&ldsH[b * 256 + 128 + c] = hc ? *(const uint4*)&hc[b * 128 + c] : z4;
    }
    __syncthreads();

    const __bf16* gxc = gx_all + (size_t)cell * 64 * 768;

    // gh = h @ W_hh^T + b_hh, fused gates. 64 (r,z,n) tile triples, 8/wave.
#pragma unroll 1
    for (int q = 0; q < 8; ++q) {
      int tau = wave * 8 + q;
      int mt = tau & 3, nt = tau >> 2;                // nt in 0..15
      v8f accR = {}, accZ = {}, accN = {};
#pragma unroll
      for (int kk = 0; kk < 256; kk += 32) {
        v16bf a  = load_a16x32(ldsH + mt * 16 * 256 + kk, 256, lane);
        v16bf br = load_b32x16(Whh_bf, 256, 0   + nt * 16, kk, lane);
        v16bf bz = load_b32x16(Whh_bf, 256, 256 + nt * 16, kk, lane);
        v16bf bn = load_b32x16(Whh_bf, 256, 512 + nt * 16, kk, lane);
        accR = wmma_bf16(a, br, accR);
        accZ = wmma_bf16(a, bz, accZ);
        accN = wmma_bf16(a, bn, accN);
      }
      int row0 = mt * 16 + (lane >> 4) * 8;
      int colr = nt * 16 + (lane & 15);               // 0..255 in GRU state
      float bhr = b_hh[colr], bhz = b_hh[256 + colr], bhn = b_hh[512 + colr];
#pragma unroll
      for (int v = 0; v < 8; ++v) {
        int b = row0 + v;
        float xr = (float)gxc[b * 768 + colr];
        float xz = (float)gxc[b * 768 + 256 + colr];
        float xn = (float)gxc[b * 768 + 512 + colr];
        float r = sigmoidf_(xr + accR[v] + bhr);
        float z = sigmoidf_(xz + accZ[v] + bhz);
        float n = tanhf(xn + r * (accN[v] + bhn));
        float h = (float)ldsH[b * 256 + colr];
        ldsHN[b * 256 + colr] = (__bf16)((1.0f - z) * n + z * h);
      }
    }
    __syncthreads();

    // h_c = h_new @ Wc^T + bc : M=64 N=128 K=256 -> 32 tiles, 4 per wave
    __bf16* hout = hbuf + (size_t)cell * 64 * 128;
#pragma unroll 1
    for (int q = 0; q < 4; ++q) {
      int t = wave * 4 + q;
      int mt = t & 3, nt = t >> 2;
      v8f acc = {};
#pragma unroll
      for (int kk = 0; kk < 256; kk += 32) {
        v16bf a = load_a16x32(ldsHN + mt * 16 * 256 + kk, 256, lane);
        v16bf b = load_b32x16(Wc_bf, 256, nt * 16, kk, lane);
        acc = wmma_bf16(a, b, acc);
      }
      int row0 = mt * 16 + (lane >> 4) * 8;
      int col  = nt * 16 + (lane & 15);
      float bias = bc[col];
#pragma unroll
      for (int v = 0; v < 8; ++v) {
        float hcv = acc[v] + bias;
        hout[(row0 + v) * 128 + col] = (__bf16)hcv;
        if (cell == 1023) out[(row0 + v) * 128 + col] = hcv;  // final (31,31)
      }
    }

    __threadfence();            // make hbuf stores agent-visible
    __syncthreads();
    if (threadIdx.x == 0)
      __hip_atomic_store(progress + j, i + 1, __ATOMIC_RELEASE,
                         __HIP_MEMORY_SCOPE_AGENT);
    __syncthreads();
  }
}

// ------------------------------- launcher ----------------------------------
extern "C" void kernel_launch(void* const* d_in, const int* in_sizes, int n_in,
                              void* d_out, int out_size, void* d_ws, size_t ws_size,
                              hipStream_t stream) {
  (void)in_sizes; (void)n_in; (void)out_size; (void)ws_size;
  const float* x    = (const float*)d_in[0];
  const float* We   = (const float*)d_in[1];
  const float* be   = (const float*)d_in[2];
  const float* W_ih = (const float*)d_in[3];
  const float* b_ih = (const float*)d_in[4];
  const float* W_hh = (const float*)d_in[5];
  const float* b_hh = (const float*)d_in[6];
  const float* Wc   = (const float*)d_in[7];
  const float* bc   = (const float*)d_in[8];
  float* out = (float*)d_out;
  char* ws = (char*)d_ws;

  // workspace layout (all offsets 256-B aligned)
  const size_t o_We   = 0;                                   // 128*64  bf16
  const size_t o_Wih  = o_We   + (size_t)128 * 64  * 2;      // 768*128 bf16
  const size_t o_Whh  = o_Wih  + (size_t)768 * 128 * 2;      // 768*256 bf16
  const size_t o_Wc   = o_Whh  + (size_t)768 * 256 * 2;      // 128*256 bf16
  const size_t o_prog = o_Wc   + (size_t)128 * 256 * 2;      // 32 ints
  const size_t o_hbuf = o_prog + 256;                        // 1024*64*128 bf16
  const size_t o_gx   = o_hbuf + (size_t)1024 * 64 * 128 * 2;// 1024*64*768 bf16

  __bf16* We_bf  = (__bf16*)(ws + o_We);
  __bf16* Wih_bf = (__bf16*)(ws + o_Wih);
  __bf16* Whh_bf = (__bf16*)(ws + o_Whh);
  __bf16* Wc_bf  = (__bf16*)(ws + o_Wc);
  int*    prog   = (int*)(ws + o_prog);
  __bf16* hbuf   = (__bf16*)(ws + o_hbuf);
  __bf16* gx_all = (__bf16*)(ws + o_gx);

  convert_bf16_kernel<<<(128 * 64  + 255) / 256, 256, 0, stream>>>(We,   We_bf,  128 * 64);
  convert_bf16_kernel<<<(768 * 128 + 255) / 256, 256, 0, stream>>>(W_ih, Wih_bf, 768 * 128);
  convert_bf16_kernel<<<(768 * 256 + 255) / 256, 256, 0, stream>>>(W_hh, Whh_bf, 768 * 256);
  convert_bf16_kernel<<<(128 * 256 + 255) / 256, 256, 0, stream>>>(Wc,   Wc_bf,  128 * 256);
  hipMemsetAsync(prog, 0, 32 * sizeof(int), stream);

  gx_kernel<<<1024, 256, 0, stream>>>(x, be, b_ih, We_bf, Wih_bf, gx_all);
  wavefront_kernel<<<32, 256, 0, stream>>>(gx_all, b_hh, bc, Whh_bf, Wc_bf,
                                           hbuf, prog, out);
}
